// MaskedAttentionHead_38508676776113
// MI455X (gfx1250) — matrix operand order repaired and checked by reading, compiled, not run on hardware
//
#include <hip/hip_runtime.h>
#include <hip/hip_bf16.h>

typedef _Float16 f16;
typedef __attribute__((ext_vector_type(16))) _Float16 v16h;
typedef __attribute__((ext_vector_type(8))) float v8f;
typedef __attribute__((ext_vector_type(4))) float v4f;
typedef __attribute__((ext_vector_type(4))) unsigned int v4u;
typedef __attribute__((ext_vector_type(4))) unsigned int u32x4;
typedef __attribute__((ext_vector_type(8))) int i32x8;
typedef __attribute__((ext_vector_type(4))) int i32x4;

#define B_ 8
#define S_ 2048
#define E_ 512
#define H_ 64
#define BS_ (B_ * S_)
#define SCALE_ 0.125f  // HEAD**-0.5

// Fragment bit-punning helpers (trivially-copyable members only).
union Frag {
  v16h h;
  f16 e[16];
  unsigned int u[8];
  v4u q[2];
};

// LDS strides (f16 elements). KSTR=72 is reproduced by the TDM pad feature:
// rows of 64 f16 = 32 DWORDs (pad_interval=4 -> 2^5 DWORDs), pad 4 DWORDs
// (pad_amount=3) = 8 halves.
#define KSTR 72
#define PSTR 40
#define TILE_H (32 * KSTR)                  // one K or V tile in LDS (halves)
#define WLDS (4 * TILE_H + 16 * PSTR)       // 2x(K+V) ping-pong + P, per wave

static __device__ __forceinline__ unsigned lds_off(const void* p) {
  // Generic pointers to LDS carry the byte offset in the low 32 bits
  return (unsigned)(unsigned long long)p;
}

// Issue one TDM DMA: 32x64 f16 row-major tile (row stride 64) from global to
// LDS with hardware row padding to KSTR=72 halves.  Tracked by TENSORcnt.
static __device__ __forceinline__ void tdm_load_tile(unsigned ldsaddr,
                                                     const f16* gsrc) {
  const unsigned long long ga = (unsigned long long)(const void*)gsrc;
  u32x4 g0;
  g0[0] = 1u;                                               // count=1 (valid)
  g0[1] = ldsaddr;                                          // lds_addr (bytes)
  g0[2] = (unsigned)ga;                                     // global_addr[31:0]
  g0[3] = (unsigned)((ga >> 32) & 0x1FFFFFFu) | (2u << 30); // [56:32] | type=2
  i32x8 g1;
  g1[0] = (1 << 16)    // data_size = 2 bytes (f16)
        | (1 << 20)    // pad_enable
        | (4 << 22)    // pad_interval: 32 DWORDs (one 128B row)
        | (3 << 25);   // pad_amount: 4 DWORDs (8 halves)
  g1[1] = (64 << 16);  // tensor_dim0 = 64  (bits 79:64 of group -> [63:48])
  g1[2] = (32 << 16);  // tensor_dim1 = 32  (bits 95:80)
  g1[3] = (64 << 16);  // tile_dim0 = 64    (bits 127:112)
  g1[4] = 32;          // tile_dim1 = 32, tile_dim2 = 0
  g1[5] = 64;          // tensor_dim0_stride = 64 elements (low 32)
  g1[6] = 0;           // stride hi / tensor_dim1_stride (unused for 2D)
  g1[7] = 0;
  const i32x4 z4 = {};
#if __clang_major__ >= 23
  const i32x8 z8 = {};
  __builtin_amdgcn_tensor_load_to_lds(g0, g1, z4, z4, z8, 0);
#else
  __builtin_amdgcn_tensor_load_to_lds(g0, g1, z4, z4, 0);
#endif
}

// ---------------------------------------------------------------------------
// Kernel 0: transpose+convert W[512][64] (f32) x3 into Wt[3][64][512] (f16)
// so WMMA B-fragment dword pairs are contiguous in memory.
// ---------------------------------------------------------------------------
__global__ __launch_bounds__(256) void wt_prep_kernel(
    const float* __restrict__ Wq, const float* __restrict__ Wk,
    const float* __restrict__ Wv, f16* __restrict__ Wt) {
  const int i = blockIdx.x * 256 + threadIdx.x;
  if (i >= 3 * H_ * E_) return;
  const int mat = i / (H_ * E_);
  const int rem = i - mat * (H_ * E_);
  const int n = rem >> 9;
  const int k = rem & 511;
  const float* W = (mat == 0) ? Wq : (mat == 1) ? Wk : Wv;
  Wt[i] = (f16)W[k * H_ + n];
}

// ---------------------------------------------------------------------------
// Kernel 1: QKV projection with V_WMMA_F32_16X16X32_F16 (f16 in, fp32 acc).
// ---------------------------------------------------------------------------
__global__ __launch_bounds__(128) void qkv_proj_f16_kernel(
    const float* __restrict__ x, const f16* __restrict__ Wt,
    f16* __restrict__ qkv) {
  const int lane = threadIdx.x & 31;
  const int wid = threadIdx.x >> 5;
  const int hi = lane >> 4;
  const int lo = lane & 15;
  const int t = blockIdx.x * 4 + wid;  // token tile 0..1023
  const int mat = blockIdx.y;          // 0=Q 1=K 2=V

  const f16* Wtm = Wt + (size_t)mat * H_ * E_;
  f16* out = qkv + (size_t)mat * BS_ * H_;
  const float* xrow = x + (size_t)(t * 16 + lo) * E_;

  v8f c0 = {}, c1 = {}, c2 = {}, c3 = {};
  for (int c = 0; c < 16; ++c) {
    const int kb = c * 32;
    const v4f f0 = *(const v4f*)(xrow + kb + 8 * hi);
    const v4f f1 = *(const v4f*)(xrow + kb + 8 * hi + 4);
    const v4f f2 = *(const v4f*)(xrow + kb + 16 + 8 * hi);
    const v4f f3 = *(const v4f*)(xrow + kb + 16 + 8 * hi + 4);
    Frag a;
#pragma unroll
    for (int i = 0; i < 4; ++i) {
      a.e[i] = (f16)f0[i];
      a.e[4 + i] = (f16)f1[i];
      a.e[8 + i] = (f16)f2[i];
      a.e[12 + i] = (f16)f3[i];
    }
    Frag b0, b1, b2, b3;
    const int ko = kb + 8 * hi;
#pragma unroll
    for (int j = 0; j < 8; ++j) {
      b0.u[j] = *(const unsigned int*)(Wtm + (size_t)(lo)*E_ + ko + 2 * j + ((j >> 2) << 3));
      b1.u[j] = *(const unsigned int*)(Wtm + (size_t)(16 + lo) * E_ + ko + 2 * j + ((j >> 2) << 3));
      b2.u[j] = *(const unsigned int*)(Wtm + (size_t)(32 + lo) * E_ + ko + 2 * j + ((j >> 2) << 3));
      b3.u[j] = *(const unsigned int*)(Wtm + (size_t)(48 + lo) * E_ + ko + 2 * j + ((j >> 2) << 3));
    }
    c0 = __builtin_amdgcn_wmma_f32_16x16x32_f16(false, a.h, false, b0.h, (short)0, c0, false, false);
    c1 = __builtin_amdgcn_wmma_f32_16x16x32_f16(false, a.h, false, b1.h, (short)0, c1, false, false);
    c2 = __builtin_amdgcn_wmma_f32_16x16x32_f16(false, a.h, false, b2.h, (short)0, c2, false, false);
    c3 = __builtin_amdgcn_wmma_f32_16x16x32_f16(false, a.h, false, b3.h, (short)0, c3, false, false);
  }
#pragma unroll
  for (int r = 0; r < 8; ++r) {
    const int m = r + 8 * hi;
    f16* orow = out + (size_t)(t * 16 + m) * H_;
    orow[lo] = (f16)c0[r];
    orow[16 + lo] = (f16)c1[r];
    orow[32 + lo] = (f16)c2[r];
    orow[48 + lo] = (f16)c3[r];
  }
}

// ---------------------------------------------------------------------------
// Kernel 2: flash attention.  2 waves/block; per wave: TDM double-buffered
// K/V tiles (DMA overlapped with compute via TENSORcnt), f16 WMMA, fp32
// online softmax, O += P V.
// ---------------------------------------------------------------------------
__global__ __launch_bounds__(64) void flash_attn_f16_kernel(
    const f16* __restrict__ qh, const f16* __restrict__ kh,
    const f16* __restrict__ vh, const int* __restrict__ mask,
    float* __restrict__ out) {
  __shared__ f16 smem[2][WLDS];
  const int lane = threadIdx.x & 31;
  const int wid = threadIdx.x >> 5;
  const int hi = lane >> 4;
  const int lo = lane & 15;

  f16* base = smem[wid];
  f16* bufK[2] = {base, base + 2 * TILE_H};
  f16* bufV[2] = {base + TILE_H, base + 3 * TILE_H};
  f16* ldsP = base + 4 * TILE_H;
  const unsigned offK[2] = {lds_off(bufK[0]), lds_off(bufK[1])};
  const unsigned offV[2] = {lds_off(bufV[0]), lds_off(bufV[1])};

  const int g = blockIdx.x * 2 + wid;  // 0..1023
  const int b = g >> 7;
  const int qt = g & 127;
  const int q0 = qt * 16;

  const f16* Qb = qh + (size_t)b * S_ * H_;
  const f16* Kb = kh + (size_t)b * S_ * H_;
  const f16* Vb = vh + (size_t)b * S_ * H_;
  const float negInf = -__builtin_huge_valf();

  // Q tile -> 2 A-fragments (K-dim 64 = 2 chunks of 32)
  Frag qa[2];
  const f16* qrow = Qb + (size_t)(q0 + lo) * H_;
#pragma unroll
  for (int c = 0; c < 2; ++c) {
    qa[c].q[0] = *(const v4u*)(qrow + c * 32 + 8 * hi);
    qa[c].q[1] = *(const v4u*)(qrow + c * 32 + 16 + 8 * hi);
  }

  v8f o0 = {}, o1 = {}, o2 = {}, o3 = {};
  float mrow[8], lrow[8], alpha[8];
#pragma unroll
  for (int r = 0; r < 8; ++r) { mrow[r] = negInf; lrow[r] = 0.f; }

  const int nb = (qt >> 1) + 1;  // 32-key blocks (causal)

  // Preload block 0 via TDM.
  tdm_load_tile(offK[0], Kb);
  tdm_load_tile(offV[0], Vb);

  for (int blk = 0; blk < nb; ++blk) {
    const int k0 = blk * 32;
    const int cur = blk & 1;
    // Kick off next block's DMA before waiting on the current one.
    if (blk + 1 < nb) {
      const int nxt = cur ^ 1;
      tdm_load_tile(offK[nxt], Kb + (size_t)(k0 + 32) * H_);
      tdm_load_tile(offV[nxt], Vb + (size_t)(k0 + 32) * H_);
      __builtin_amdgcn_s_wait_tensorcnt(2);  // current K,V complete (in-order)
    } else {
      __builtin_amdgcn_s_wait_tensorcnt(0);
    }
    asm volatile("" ::: "memory");  // keep LDS reads below the wait

    f16* ldsK = bufK[cur];
    f16* ldsV = bufV[cur];
    const int mv0 = mask[b * S_ + k0 + lo];
    const int mv1 = mask[b * S_ + k0 + 16 + lo];

    // S = Q K^T over 32 keys: two 16x16 C-frags, 2 K-chunks each.
    v8f s0f = {}, s1f = {};
#pragma unroll
    for (int c = 0; c < 2; ++c) {
      const int ko = c * 32 + 8 * hi;
      Frag bk0, bk1;
#pragma unroll
      for (int j = 0; j < 8; ++j) {
        bk0.u[j] = *(const unsigned int*)(ldsK + lo * KSTR + ko + 2 * j + ((j >> 2) << 3));
        bk1.u[j] = *(const unsigned int*)(ldsK + (16 + lo) * KSTR + ko + 2 * j + ((j >> 2) << 3));
      }
      s0f = __builtin_amdgcn_wmma_f32_16x16x32_f16(false, qa[c].h, false, bk0.h, (short)0, s0f, false, false);
      s1f = __builtin_amdgcn_wmma_f32_16x16x32_f16(false, qa[c].h, false, bk1.h, (short)0, s1f, false, false);
    }

    // fp32 online softmax over the 32-key block.
#pragma unroll
    for (int r = 0; r < 8; ++r) {
      const int m = r + 8 * hi;
      float s0 = s0f[r] * SCALE_;
      float s1 = s1f[r] * SCALE_;
      const bool keep0 = ((k0 + lo) <= (q0 + m)) && (mv0 != 0);
      const bool keep1 = ((k0 + 16 + lo) <= (q0 + m)) && (mv1 != 0);
      s0 = keep0 ? s0 : negInf;
      s1 = keep1 ? s1 : negInf;
      float rm = fmaxf(s0, s1);
      rm = fmaxf(rm, __shfl_xor(rm, 1, 32));
      rm = fmaxf(rm, __shfl_xor(rm, 2, 32));
      rm = fmaxf(rm, __shfl_xor(rm, 4, 32));
      rm = fmaxf(rm, __shfl_xor(rm, 8, 32));
      const float mn = fmaxf(mrow[r], rm);
      const float a = __expf(mrow[r] - mn);
      const float p0 = __expf(s0 - mn);
      const float p1 = __expf(s1 - mn);
      float rs = p0 + p1;
      rs += __shfl_xor(rs, 1, 32);
      rs += __shfl_xor(rs, 2, 32);
      rs += __shfl_xor(rs, 4, 32);
      rs += __shfl_xor(rs, 8, 32);
      lrow[r] = lrow[r] * a + rs;
      mrow[r] = mn;
      alpha[r] = a;
      ldsP[m * PSTR + lo] = (f16)p0;
      ldsP[m * PSTR + 16 + lo] = (f16)p1;
    }
#pragma unroll
    for (int r = 0; r < 8; ++r) {
      o0[r] *= alpha[r]; o1[r] *= alpha[r]; o2[r] *= alpha[r]; o3[r] *= alpha[r];
    }

    // O += P V (single K=32 WMMA per 16-col output tile).
    Frag pa;
#pragma unroll
    for (int j = 0; j < 8; ++j) {
      const int koff = (j < 4) ? (8 * hi + 2 * j) : (16 + 8 * hi + 2 * (j - 4));
      pa.u[j] = *(const unsigned int*)(ldsP + lo * PSTR + koff);
    }
    const unsigned short* Vu = (const unsigned short*)ldsV;
    Frag bv0, bv1, bv2, bv3;
#pragma unroll
    for (int j = 0; j < 8; ++j) {
      const int kr = 8 * hi + 2 * j + ((j >> 2) << 3);
      bv0.u[j] = (unsigned)Vu[kr * KSTR + lo] | ((unsigned)Vu[(kr + 1) * KSTR + lo] << 16);
      bv1.u[j] = (unsigned)Vu[kr * KSTR + 16 + lo] | ((unsigned)Vu[(kr + 1) * KSTR + 16 + lo] << 16);
      bv2.u[j] = (unsigned)Vu[kr * KSTR + 32 + lo] | ((unsigned)Vu[(kr + 1) * KSTR + 32 + lo] << 16);
      bv3.u[j] = (unsigned)Vu[kr * KSTR + 48 + lo] | ((unsigned)Vu[(kr + 1) * KSTR + 48 + lo] << 16);
    }
    o0 = __builtin_amdgcn_wmma_f32_16x16x32_f16(false, pa.h, false, bv0.h, (short)0, o0, false, false);
    o1 = __builtin_amdgcn_wmma_f32_16x16x32_f16(false, pa.h, false, bv1.h, (short)0, o1, false, false);
    o2 = __builtin_amdgcn_wmma_f32_16x16x32_f16(false, pa.h, false, bv2.h, (short)0, o2, false, false);
    o3 = __builtin_amdgcn_wmma_f32_16x16x32_f16(false, pa.h, false, bv3.h, (short)0, o3, false, false);
  }

  // Epilogue: normalize by fp32 row sums, write fp32 16x64 tile.
#pragma unroll
  for (int r = 0; r < 8; ++r) {
    const int m = r + 8 * hi;
    const float inv = 1.0f / lrow[r];
    float* orow = out + (size_t)b * S_ * H_ + (size_t)(q0 + m) * H_;
    orow[lo] = o0[r] * inv;
    orow[16 + lo] = o1[r] * inv;
    orow[32 + lo] = o2[r] * inv;
    orow[48 + lo] = o3[r] * inv;
  }
}

extern "C" void kernel_launch(void* const* d_in, const int* in_sizes, int n_in,
                              void* d_out, int out_size, void* d_ws, size_t ws_size,
                              hipStream_t stream) {
  const float* x = (const float*)d_in[0];
  const float* Wq = (const float*)d_in[1];
  const float* Wk = (const float*)d_in[2];
  const float* Wv = (const float*)d_in[3];
  const int* mask = (const int*)d_in[4];
  float* out = (float*)d_out;

  // Workspace (f16): Wt[3][64][512] then QKV[3][16384][64]  (~6.5 MB)
  f16* Wt = (f16*)d_ws;
  f16* qkv = Wt + (size_t)3 * H_ * E_;

  wt_prep_kernel<<<(3 * H_ * E_ + 255) / 256, 256, 0, stream>>>(Wq, Wk, Wv, Wt);

  dim3 g1((BS_ / 16) / 4, 3);
  qkv_proj_f16_kernel<<<g1, 128, 0, stream>>>(x, Wt, qkv);

  const f16* qh = qkv;
  const f16* kh = qkv + (size_t)BS_ * H_;
  const f16* vh = qkv + (size_t)2 * BS_ * H_;
  flash_attn_f16_kernel<<<(BS_ / 16) / 2, 64, 0, stream>>>(qh, kh, vh, mask, out);
}
